// NodeGLAM_53257594470865
// MI455X (gfx1250) — matrix operand order repaired and checked by reading, not compiled, not run on hardware
//
#include <hip/hip_runtime.h>
#include <hip/hip_bf16.h>
#include <math.h>

#define N_NODES 100000
#define N_EDGES 1000000
#define BN_EPS 1e-5f

typedef __bf16 bf16;
typedef __attribute__((ext_vector_type(16))) __bf16 v16bf;
typedef __attribute__((ext_vector_type(8)))  __bf16 v8bf;
typedef __attribute__((ext_vector_type(8)))  float  v8f;

__device__ __forceinline__ float gelu_erf(float v) {
  return 0.5f * v * (1.0f + erff(v * 0.7071067811865475f));
}

// ---------------- graph normalization ----------------
__global__ void k_degree(const long long* __restrict__ dst, float* deg, int E) {
  int e = blockIdx.x * blockDim.x + threadIdx.x;
  if (e < E)
    __hip_atomic_fetch_add(&deg[(int)dst[e]], 1.0f, __ATOMIC_RELAXED, __HIP_MEMORY_SCOPE_AGENT);
}

__global__ void k_dis(float* deg, int n) {
  int i = blockIdx.x * blockDim.x + threadIdx.x;
  if (i < n) {
    float d = deg[i];
    deg[i] = (d > 0.0f) ? rsqrtf(fmaxf(d, 1.0f)) : 0.0f;
  }
}

__global__ void k_edgenorm(const long long* __restrict__ src, const long long* __restrict__ dst,
                           const float* __restrict__ dis, float* enorm, int E) {
  int e = blockIdx.x * blockDim.x + threadIdx.x;
  if (e < E) enorm[e] = dis[(int)src[e]] * dis[(int)dst[e]];
}

// ---------------- batchnorm ----------------
__global__ void k_bn_stats(const float* __restrict__ x, float* colsum, float* colss, int rows) {
  int col = threadIdx.x;          // 128 threads
  int r0 = blockIdx.x * 64;
  int r1 = min(r0 + 64, rows);
  float s = 0.f, ss = 0.f;
  for (int r = r0; r < r1; ++r) {
    float v = x[(size_t)r * 128 + col];
    s += v; ss += v * v;
  }
  __hip_atomic_fetch_add(&colsum[col], s,  __ATOMIC_RELAXED, __HIP_MEMORY_SCOPE_AGENT);
  __hip_atomic_fetch_add(&colss[col],  ss, __ATOMIC_RELAXED, __HIP_MEMORY_SCOPE_AGENT);
}

__global__ void k_bn_finalize(float* colsum, float* colss) {
  int c = threadIdx.x;            // 128 threads, 1 block
  float mean = colsum[c] * (1.0f / (float)N_NODES);
  float var  = colss[c]  * (1.0f / (float)N_NODES) - mean * mean;
  colsum[c] = mean;
  colss[c]  = rsqrtf(var + BN_EPS);
}

// normalize x -> bf16 (GEMM1 input) and copy raw x -> bf16 into concat buffer cols 0..127
__global__ void k_bn_apply(const float* __restrict__ x, const float* __restrict__ mean,
                           const float* __restrict__ invstd, const float* __restrict__ gamma,
                           const float* __restrict__ beta, bf16* xnb, bf16* catb) {
  int t = blockIdx.x * blockDim.x + threadIdx.x;
  if (t >= N_NODES * 128) return;
  int col = t & 127;
  int row = t >> 7;
  float v = x[t];
  float nv = (v - mean[col]) * invstd[col] * gamma[col] + beta[col];
  xnb[t] = (bf16)nv;
  catb[(size_t)row * 192 + col] = (bf16)v;
}

// copy fp32 [rows x 64] (ld 64) into bf16 dst with leading dim ldd at column offset coff
__global__ void k_cvt_cols(const float* __restrict__ s, bf16* d, int ldd, int coff) {
  int t = blockIdx.x * blockDim.x + threadIdx.x;
  if (t >= N_NODES * 64) return;
  int col = t & 63;
  int row = t >> 6;
  d[(size_t)row * ldd + coff + col] = (bf16)s[t];
}

// ---------------- pack weight matrix into WMMA B-fragment layout (fused f32->bf16) ----
// Fragment order: [kc][nt][lane 0..31][elem 0..15], 16 contiguous bf16 (32 B) per lane.
// k contiguous: lane<16 -> N=nt*16+lane,    elem i holds K = kc*32 + i
//               lane>=16 -> N=nt*16+lane-16, elem i holds K = kc*32 + 16 + i
__global__ void k_packB(const float* __restrict__ W, bf16* __restrict__ P, int Kdim, int N) {
  int t = blockIdx.x * blockDim.x + threadIdx.x;
  if (t >= Kdim * N) return;
  int i    = t & 15;
  int lane = (t >> 4) & 31;
  int frag = t >> 9;
  int ntN  = N >> 4;
  int nt   = frag % ntN;
  int kc   = frag / ntN;
  int k = (kc << 5) + ((lane >> 4) << 4) + i;
  int n = (nt << 4) + (lane & 15);
  P[t] = (bf16)W[(size_t)k * N + n];
}

// ---------------- edge propagation: out[dst] += norm * in[src], feat dim 64 ----------------
__global__ void k_prop(const long long* __restrict__ src, const long long* __restrict__ dst,
                       const float* __restrict__ enorm,
                       const float* __restrict__ hin, float* hout, int E) {
  int t = blockIdx.x * blockDim.x + threadIdx.x;
  int e = t >> 4;               // 16 lanes per edge
  if (e >= E) return;
  int lane = t & 15;            // 4 floats per lane
  int s = (int)src[e];
  int d = (int)dst[e];
  float w = enorm[e];
  const float4 v = *(const float4*)(hin + (size_t)s * 64 + lane * 4);
  float* o = hout + (size_t)d * 64 + lane * 4;
  __hip_atomic_fetch_add(o + 0, v.x * w, __ATOMIC_RELAXED, __HIP_MEMORY_SCOPE_AGENT);
  __hip_atomic_fetch_add(o + 1, v.y * w, __ATOMIC_RELAXED, __HIP_MEMORY_SCOPE_AGENT);
  __hip_atomic_fetch_add(o + 2, v.z * w, __ATOMIC_RELAXED, __HIP_MEMORY_SCOPE_AGENT);
  __hip_atomic_fetch_add(o + 3, v.w * w, __ATOMIC_RELAXED, __HIP_MEMORY_SCOPE_AGENT);
}

// ---------------- bf16 WMMA GEMM: one wave per 16-row strip, NT col tiles ----------------
// C[16 x NT*16] = A[16 x K] @ Bpacked + bias ; optional erf-GELU ; fp32 and/or bf16 out
template <int NT>
__global__ void k_gemm_bf16(const bf16* __restrict__ A, int lda,
                            const bf16* __restrict__ Bp,
                            const float* __restrict__ bias,
                            float* outF, int ldF,
                            bf16* outB, int ldB, int coffB,
                            int Mtiles, int K, int do_gelu) {
  const int wave = threadIdx.x >> 5;
  const int lane = threadIdx.x & 31;
  int strip = blockIdx.x * (blockDim.x >> 5) + wave;
  if (strip >= Mtiles) return;         // wave-uniform exit: EXEC all-1s for WMMA
  const int row0 = strip << 4;

  // A fragment (contiguous k order, same as packed B): lane<16 -> M=lane, K = +0..15
  //                                     lane>=16 -> M=lane-16, K = +16..31
  const int am  = lane & 15;
  const int akb = (lane >> 4) << 4;    // 0 or 16
  const bf16* arow = A + (size_t)(row0 + am) * lda + akb;
  const bf16* bfragp = Bp + (size_t)lane * 16;   // lane's slot inside fragment 0

  v8f acc[NT];
#pragma unroll
  for (int nt = 0; nt < NT; ++nt) acc[nt] = (v8f){};

  const int ntK = K >> 5;
  for (int kc = 0; kc < ntK; ++kc) {
    v8bf lo = *(const v8bf*)(arow + (kc << 5));
    v8bf hi = *(const v8bf*)(arow + (kc << 5) + 8);
    v16bf a = __builtin_shufflevector(lo, hi, 0, 1, 2, 3, 4, 5, 6, 7,
                                      8, 9, 10, 11, 12, 13, 14, 15);
#pragma unroll
    for (int nt = 0; nt < NT; ++nt) {
      // fragment (kc, nt): 32 lanes x 16 bf16, contiguous
      v16bf b = *(const v16bf*)(bfragp + ((size_t)(kc * NT + nt) << 9));
      acc[nt] = __builtin_amdgcn_wmma_f32_16x16x32_bf16(false, a, false, b, (short)0,
                                                        acc[nt], false, false);
    }
  }

  // C/D layout: lane<16 -> N=lane, M=r ; lane>=16 -> N=lane-16, M=8+r
  const int on  = lane & 15;
  const int omb = (lane >> 4) << 3;
#pragma unroll
  for (int nt = 0; nt < NT; ++nt) {
    const int col = (nt << 4) + on;
    const float bv = bias ? bias[col] : 0.0f;
#pragma unroll
    for (int r = 0; r < 8; ++r) {
      float v = acc[nt][r] + bv;
      if (do_gelu) v = gelu_erf(v);
      const int m = row0 + omb + r;
      if (outF) outF[(size_t)m * ldF + col] = v;
      if (outB) outB[(size_t)m * ldB + coffB + col] = (bf16)v;
    }
  }
}

// ---------------- host ----------------
static void launch_gemm(const bf16* A, int lda, const bf16* Bp, const float* bias,
                        float* outF, int ldF, bf16* outB, int ldB, int coffB,
                        int M, int N, int K, int g, hipStream_t s) {
  int Mtiles = M / 16;
  int blocks = (Mtiles + 7) / 8;        // 8 waves (row strips) per 256-thread block
  if (N == 64)
    k_gemm_bf16<4><<<blocks, 256, 0, s>>>(A, lda, Bp, bias, outF, ldF, outB, ldB, coffB,
                                          Mtiles, K, g);
  else // N == 128
    k_gemm_bf16<8><<<blocks, 256, 0, s>>>(A, lda, Bp, bias, outF, ldF, outB, ldB, coffB,
                                          Mtiles, K, g);
}

extern "C" void kernel_launch(void* const* d_in, const int* in_sizes, int n_in,
                              void* d_out, int out_size, void* d_ws, size_t ws_size,
                              hipStream_t stream) {
  const float*     x     = (const float*)d_in[0];
  const long long* ei    = (const long long*)d_in[1];   // int64 [2, E]
  const long long* esrc  = ei;
  const long long* edst  = ei + N_EDGES;
  const float* gamma = (const float*)d_in[2];
  const float* beta  = (const float*)d_in[3];
  const float* W1    = (const float*)d_in[4];
  const float* b1    = (const float*)d_in[5];
  const float* tag1W = (const float*)d_in[6];   // [4,64,64] == row-major 256x64
  const float* tag1b = (const float*)d_in[7];
  const float* W2    = (const float*)d_in[8];
  const float* b2    = (const float*)d_in[9];
  const float* tag2W = (const float*)d_in[10];
  const float* tag2b = (const float*)d_in[11];
  const float* W3    = (const float*)d_in[12];  // [192,128]
  const float* b3    = (const float*)d_in[13];
  const float* W4    = (const float*)d_in[14];  // [128,64]
  const float* b4    = (const float*)d_in[15];
  float* out = (float*)d_out;                   // [100000, 64] fp32

  char* wsb = (char*)d_ws;
  size_t off = 0;
  auto alloc = [&](size_t bytes) -> char* {
    char* p = wsb + off;
    off = (off + bytes + 255) & ~(size_t)255;
    return p;
  };
  float* deg    = (float*)alloc((size_t)N_NODES * 4);           // reused as dis
  float* enorm  = (float*)alloc((size_t)N_EDGES * 4);
  float* colsum = (float*)alloc(128 * 4);
  float* colss  = (float*)alloc(128 * 4);
  bf16*  xnb    = (bf16*)alloc((size_t)N_NODES * 128 * 2);      // BN(x) bf16
  bf16*  hcat   = (bf16*)alloc((size_t)N_NODES * 256 * 2);      // [h0|h1|h2|h3]
  bf16*  catb   = (bf16*)alloc((size_t)N_NODES * 192 * 2);      // [x | tag2 out]
  bf16*  t64b   = (bf16*)alloc((size_t)N_NODES * 64 * 2);
  float* h32a   = (float*)alloc((size_t)N_NODES * 64 * 4);
  float* h32b   = (float*)alloc((size_t)N_NODES * 64 * 4);
  bf16*  W1p    = (bf16*)alloc(128 * 64 * 2);   // packed B fragments
  bf16*  T1p    = (bf16*)alloc(256 * 64 * 2);
  bf16*  W2p    = (bf16*)alloc(64 * 64 * 2);
  bf16*  T2p    = (bf16*)alloc(256 * 64 * 2);
  bf16*  W3p    = (bf16*)alloc(192 * 128 * 2);
  bf16*  W4p    = (bf16*)alloc(128 * 64 * 2);
  bf16*  ab16   = hcat;   // alias: hcat dead after TAG2 GEMM; reuse for 'a' activation

  // ---- graph norm ----
  hipMemsetAsync(deg, 0, (size_t)N_NODES * 4, stream);
  k_degree<<<(N_EDGES + 255) / 256, 256, 0, stream>>>(edst, deg, N_EDGES);
  k_dis<<<(N_NODES + 255) / 256, 256, 0, stream>>>(deg, N_NODES);
  k_edgenorm<<<(N_EDGES + 255) / 256, 256, 0, stream>>>(esrc, edst, deg, enorm, N_EDGES);

  // ---- batchnorm ----
  hipMemsetAsync(colsum, 0, 128 * 4, stream);
  hipMemsetAsync(colss,  0, 128 * 4, stream);
  k_bn_stats<<<(N_NODES + 63) / 64, 128, 0, stream>>>(x, colsum, colss, N_NODES);
  k_bn_finalize<<<1, 128, 0, stream>>>(colsum, colss);
  k_bn_apply<<<(N_NODES * 128 + 255) / 256, 256, 0, stream>>>(x, colsum, colss, gamma, beta,
                                                              xnb, catb);

  // ---- weight pack fp32 -> bf16 WMMA fragments ----
  k_packB<<<(128 * 64 + 255) / 256, 256, 0, stream>>>(W1, W1p, 128, 64);
  k_packB<<<(256 * 64 + 255) / 256, 256, 0, stream>>>(tag1W, T1p, 256, 64);
  k_packB<<<(64 * 64 + 255) / 256, 256, 0, stream>>>(W2, W2p, 64, 64);
  k_packB<<<(256 * 64 + 255) / 256, 256, 0, stream>>>(tag2W, T2p, 256, 64);
  k_packB<<<(192 * 128 + 255) / 256, 256, 0, stream>>>(W3, W3p, 192, 128);
  k_packB<<<(128 * 64 + 255) / 256, 256, 0, stream>>>(W4, W4p, 128, 64);

  // ---- layer 1: h0 = gelu(BN(x) @ W1 + b1)  -> hcat[:,0:64] (bf16) + h32a (fp32) ----
  launch_gemm(xnb, 128, W1p, b1, h32a, 64, hcat, 256, 0, N_NODES, 64, 128, 1, stream);

  // ---- TAGConv1 hops ----
  for (int k = 1; k <= 3; ++k) {
    hipMemsetAsync(h32b, 0, (size_t)N_NODES * 64 * 4, stream);
    k_prop<<<(N_EDGES * 16 + 255) / 256, 256, 0, stream>>>(esrc, edst, enorm, h32a, h32b,
                                                           N_EDGES);
    k_cvt_cols<<<(N_NODES * 64 + 255) / 256, 256, 0, stream>>>(h32b, hcat, 256, 64 * k);
    float* t = h32a; h32a = h32b; h32b = t;
  }
  // out = gelu(hcat @ stack(tag1_W) + tag1_b)
  launch_gemm(hcat, 256, T1p, tag1b, nullptr, 0, t64b, 64, 0, N_NODES, 64, 256, 1, stream);

  // ---- layer 2: gelu(h @ W2 + b2) -> hcat[:,0:64] + h32a ----
  launch_gemm(t64b, 64, W2p, b2, h32a, 64, hcat, 256, 0, N_NODES, 64, 64, 1, stream);

  // ---- TAGConv2 hops ----
  for (int k = 1; k <= 3; ++k) {
    hipMemsetAsync(h32b, 0, (size_t)N_NODES * 64 * 4, stream);
    k_prop<<<(N_EDGES * 16 + 255) / 256, 256, 0, stream>>>(esrc, edst, enorm, h32a, h32b,
                                                           N_EDGES);
    k_cvt_cols<<<(N_NODES * 64 + 255) / 256, 256, 0, stream>>>(h32b, hcat, 256, 64 * k);
    float* t = h32a; h32a = h32b; h32b = t;
  }
  // gelu(hcat @ stack(tag2_W) + tag2_b) -> catb[:,128:192]
  launch_gemm(hcat, 256, T2p, tag2b, nullptr, 0, catb, 192, 128, N_NODES, 64, 256, 1, stream);

  // ---- a = gelu([x|h] @ W3 + b3) -> ab16 (aliases hcat) ----
  launch_gemm(catb, 192, W3p, b3, nullptr, 0, ab16, 128, 0, N_NODES, 128, 192, 1, stream);

  // ---- out = a @ W4 + b4 (fp32 to d_out) ----
  launch_gemm(ab16, 128, W4p, b4, out, 64, nullptr, 0, 0, N_NODES, 64, 128, 0, stream);
}
